// CausalSelfAttention_20753281975007
// MI455X (gfx1250) — matrix operand order repaired and checked by reading, not compiled, not run on hardware
//
#include <hip/hip_runtime.h>
#include <hip/hip_bf16.h>

typedef __bf16 bf16_t;
typedef __attribute__((ext_vector_type(16))) __bf16 v16bf;
typedef __attribute__((ext_vector_type(8)))  float  v8f;
typedef __attribute__((ext_vector_type(4)))  unsigned int v4u;
typedef __attribute__((ext_vector_type(4)))  int v4i;
typedef __attribute__((ext_vector_type(8)))  int v8i;
typedef unsigned int u32;

#define BB 2
#define TT 4096
#define NH 12
#define HD 64
#define CE 768

#if defined(__has_builtin)
#if __has_builtin(__builtin_amdgcn_tensor_load_to_lds)
#define HAVE_TDM 1
#endif
#endif
#ifndef HAVE_TDM
#define HAVE_TDM 0
#endif
#if HAVE_TDM && defined(__clang_major__) && (__clang_major__ >= 23)
#define TDM_SIX_ARGS 1
#else
#define TDM_SIX_ARGS 0
#endif

static __device__ __forceinline__ v8f wmma_bf16(v16bf a, v16bf b, v8f c) {
  return __builtin_amdgcn_wmma_f32_16x16x32_bf16(false, a, false, b, (short)0, c, false, false);
}

// A-fragment (16x32 bf16, MxK): lane m=lane&15, h=lane>>4 holds K=8h..8h+7 and 16+8h..+7
static __device__ __forceinline__ v16bf frag_from_lds_a(const bf16_t* tile, int stride) {
  const int lane = threadIdx.x & 31;
  const int m = lane & 15, h = lane >> 4;
  const bf16_t* p = tile + m * stride + 8 * h;
  union { v16bf v; uint4 u[2]; } f;
  f.u[0] = *(const uint4*)(p);
  f.u[1] = *(const uint4*)(p + 16);
  return f.v;
}

// B-fragment (32x16 bf16, KxN) from K-major tile rows=N, cols=K: lane n holds K=8h..8h+7 and 16+8h..+7
static __device__ __forceinline__ v16bf frag_from_lds_b(const bf16_t* tile, int stride) {
  const int lane = threadIdx.x & 31;
  const int n = lane & 15, h = lane >> 4;
  const bf16_t* p = tile + n * stride + 8 * h;
  union { v16bf v; uint4 u[2]; } f;
  f.u[0] = *(const uint4*)(p);
  f.u[1] = *(const uint4*)(p + 16);
  return f.v;
}

#if HAVE_TDM
// One TDM 2D tile load (global -> LDS, hardware pad insertion for LDS stride).
// D# packing per CDNA5 ISA ch.8. All dims/strides are literals at call sites,
// so only global_addr costs SALU per iteration.
static __device__ __forceinline__ void tdm_load_2d(u32 lds_addr, const void* gaddr,
                                                   u32 flags_d0, u32 tdim0, u32 tdim1,
                                                   u32 tile0, u32 tile1, u32 stride0) {
  const unsigned long long ga = (unsigned long long)gaddr;
  v4u g0;
  g0.x = 1u;                              // count=1, user mode
  g0.y = lds_addr;                        // LDS byte address
  g0.z = (u32)ga;                         // global_addr[31:0]
  g0.w = (u32)(ga >> 32) | (2u << 30);    // global_addr[56:32] | type=2
  v8i g1 = {
    (int)flags_d0,                                          // mask=0 | data_size | pad cfg
    (int)((tdim0 & 0xffffu) << 16),                         // atomic_addr=0 | tdim0 lo
    (int)((tdim0 >> 16) | ((tdim1 & 0xffffu) << 16)),       // tdim0 hi | tdim1 lo
    (int)((tdim1 >> 16) | (tile0 << 16)),                   // tdim1 hi | tile_dim0
    (int)tile1,                                             // tile_dim1 | tile_dim2=0
    (int)stride0,                                           // tensor_dim0_stride lo
    0, 0
  };
  v4i z4 = {0, 0, 0, 0};
#if TDM_SIX_ARGS
  v8i z8 = {0, 0, 0, 0, 0, 0, 0, 0};
  __builtin_amdgcn_tensor_load_to_lds(g0, g1, z4, z4, z8, 0);
#else
  __builtin_amdgcn_tensor_load_to_lds(g0, g1, z4, z4, 0);
#endif
}
// flags: data_size=2B (1<<16) | pad_enable (1<<20) | pad_interval<<22 | pad_amount<<25
#define TDM_FLAGS_K ((1u << 16) | (1u << 20) | (4u << 22) | (7u << 25)) // 32dw rows +8dw pad
#define TDM_FLAGS_V ((1u << 16) | (1u << 20) | (3u << 22) | (3u << 25)) // 16dw rows +4dw pad
#endif

// ---------------------------------------------------------------------------
// Split-bf16 GEMM: C[64x64 tile] = A(fp32) * B(fp32) + bias, via hi/lo bf16x3.
// MODE 0: QKV epilogue (q,k hi/lo -> [B,H,T,D]; v -> [B,H,D,T] transposed)
// MODE 1: output projection epilogue (fp32 store)
// ---------------------------------------------------------------------------
template<int MODE>
__global__ __launch_bounds__(128)
void gemm_split(const float* __restrict__ A, const float* __restrict__ Bm,
                const float* __restrict__ bias,
                bf16_t* __restrict__ qh, bf16_t* __restrict__ ql,
                bf16_t* __restrict__ kh, bf16_t* __restrict__ kl,
                bf16_t* __restrict__ vv, float* __restrict__ outf,
                int Kdim, int ldB)
{
  __shared__ __align__(16) bf16_t Ah[64 * 40];
  __shared__ __align__(16) bf16_t Al[64 * 40];
  __shared__ __align__(16) bf16_t Bh[64 * 40];
  __shared__ __align__(16) bf16_t Bl[64 * 40];

  const int tid  = threadIdx.x;
  const int lane = tid & 31;
  const int w    = tid >> 5;
  const int row0 = blockIdx.x * 64;
  const int n0   = blockIdx.y * 64;

  v8f acc[4] = {};

  for (int k0 = 0; k0 < Kdim; k0 += 32) {
    {
      const int r  = tid >> 1;
      const int c0 = (tid & 1) * 16;
      const float* src = A + (size_t)(row0 + r) * Kdim + k0 + c0;
      if (k0 + 32 < Kdim) __builtin_prefetch(src + 32, 0, 1);
#pragma unroll
      for (int i = 0; i < 16; i += 4) {
        float4 f = *(const float4*)(src + i);
        float fs[4] = {f.x, f.y, f.z, f.w};
#pragma unroll
        for (int j = 0; j < 4; ++j) {
          bf16_t hi = (bf16_t)fs[j];
          bf16_t lo = (bf16_t)(fs[j] - (float)hi);
          Ah[r * 40 + c0 + i + j] = hi;
          Al[r * 40 + c0 + i + j] = lo;
        }
      }
    }
    {
      const int kk = tid >> 2;
      const int c0 = (tid & 3) * 16;
      const float* src = Bm + (size_t)(k0 + kk) * ldB + n0 + c0;
      if (k0 + 32 < Kdim) __builtin_prefetch(src + 32 * ldB, 0, 1);
#pragma unroll
      for (int i = 0; i < 16; i += 4) {
        float4 f = *(const float4*)(src + i);
        float fs[4] = {f.x, f.y, f.z, f.w};
#pragma unroll
        for (int j = 0; j < 4; ++j) {
          bf16_t hi = (bf16_t)fs[j];
          bf16_t lo = (bf16_t)(fs[j] - (float)hi);
          Bh[(c0 + i + j) * 40 + kk] = hi;
          Bl[(c0 + i + j) * 40 + kk] = lo;
        }
      }
    }
    __syncthreads();

    v16bf a_h = frag_from_lds_a(Ah + (16 * w) * 40, 40);
    v16bf a_l = frag_from_lds_a(Al + (16 * w) * 40, 40);
#pragma unroll
    for (int nt = 0; nt < 4; ++nt) {
      v16bf b_h = frag_from_lds_b(Bh + (16 * nt) * 40, 40);
      v16bf b_l = frag_from_lds_b(Bl + (16 * nt) * 40, 40);
      acc[nt] = wmma_bf16(a_h, b_h, acc[nt]);
      acc[nt] = wmma_bf16(a_h, b_l, acc[nt]);
      acc[nt] = wmma_bf16(a_l, b_h, acc[nt]);
    }
    __syncthreads();
  }

  const int h2 = lane >> 4, nn = lane & 15;
#pragma unroll
  for (int nt = 0; nt < 4; ++nt) {
    const int gn = n0 + nt * 16 + nn;
    const float bv = bias[gn];
#pragma unroll
    for (int r = 0; r < 8; ++r) {
      const int gm = row0 + 16 * w + 8 * h2 + r;
      const float val = acc[nt][r] + bv;
      if (MODE == 0) {
        const int which = gn / CE;          // 0=q, 1=k, 2=v
        const int c = gn % CE;
        const int hh = c >> 6, d = c & 63;
        const int bidx = gm >> 12, t = gm & (TT - 1);
        if (which == 0) {
          const size_t idx = ((size_t)(bidx * NH + hh) * TT + t) * HD + d;
          bf16_t hi = (bf16_t)val; qh[idx] = hi; ql[idx] = (bf16_t)(val - (float)hi);
        } else if (which == 1) {
          const size_t idx = ((size_t)(bidx * NH + hh) * TT + t) * HD + d;
          bf16_t hi = (bf16_t)val; kh[idx] = hi; kl[idx] = (bf16_t)(val - (float)hi);
        } else {
          // v stored transposed [B,H,D,T] so attention can TDM-load it tiled
          const size_t idx = ((size_t)(bidx * NH + hh) * HD + d) * TT + t;
          vv[idx] = (bf16_t)val;
        }
      } else {
        outf[(size_t)gm * CE + gn] = val;
      }
    }
  }
}

// ---------------------------------------------------------------------------
// Flash-style causal attention with double-buffered TDM pipeline:
//   wave0 issues tensor_load_to_lds for tile j+1 while all waves run the
//   WMMA/softmax for tile j; one barrier per k-tile.
// ---------------------------------------------------------------------------
__global__ __launch_bounds__(128)
void attn_flash(const bf16_t* __restrict__ qhg, const bf16_t* __restrict__ qlg,
                const bf16_t* __restrict__ khg, const bf16_t* __restrict__ klg,
                const bf16_t* __restrict__ vvg, float* __restrict__ att)
{
  __shared__ __align__(16) bf16_t KhB[2][32 * 80];
  __shared__ __align__(16) bf16_t KlB[2][32 * 80];
  __shared__ __align__(16) bf16_t VtB[2][64 * 40];   // rows = d, cols = k
  __shared__ __align__(16) bf16_t Pb[4 * 16 * 40];

  const int tid = threadIdx.x, lane = tid & 31, w = tid >> 5;
  const int q0 = blockIdx.x * 64;
  const int bh = blockIdx.y;
  const int b = bh / NH, h = bh % NH;
  const size_t hb = (size_t)bh * TT * HD;

  const int nn = lane & 15, h2 = lane >> 4;

  v16bf qfh[2], qfl[2];
  {
    const int qrow = q0 + 16 * w + nn;
#pragma unroll
    for (int ds = 0; ds < 2; ++ds) {
      const bf16_t* p0 = qhg + hb + (size_t)qrow * HD + ds * 32 + 8 * h2;
      union { v16bf v; uint4 u[2]; } f;
      f.u[0] = *(const uint4*)p0; f.u[1] = *(const uint4*)(p0 + 16);
      qfh[ds] = f.v;
      const bf16_t* p1 = qlg + hb + (size_t)qrow * HD + ds * 32 + 8 * h2;
      union { v16bf v; uint4 u[2]; } g;
      g.u[0] = *(const uint4*)p1; g.u[1] = *(const uint4*)(p1 + 16);
      qfl[ds] = g.v;
    }
  }

  v8f O[4] = {};
  float runmax[8], lsum[8];
#pragma unroll
  for (int r = 0; r < 8; ++r) { runmax[r] = -1e30f; lsum[r] = 0.f; }

  bf16_t* Pw = Pb + w * (16 * 40);
  const int qmax = q0 + 16 * w + 15;
  const int jend = (q0 + 64) / 32;

#if HAVE_TDM
  if (w == 0) {   // prologue: fetch tile 0 into buffer 0
    tdm_load_2d((u32)(uintptr_t)&KhB[0][0], khg + hb, TDM_FLAGS_K, 64u, (u32)TT, 64u, 32u, 64u);
    tdm_load_2d((u32)(uintptr_t)&KlB[0][0], klg + hb, TDM_FLAGS_K, 64u, (u32)TT, 64u, 32u, 64u);
    tdm_load_2d((u32)(uintptr_t)&VtB[0][0], vvg + hb, TDM_FLAGS_V, 32u, (u32)HD, 32u, 64u, (u32)TT);
  }
#endif

  for (int j = 0; j < jend; ++j) {
    const int kb = j * 32;
    const int buf = j & 1;
    const bf16_t* Kh = KhB[buf];
    const bf16_t* Kl = KlB[buf];
    const bf16_t* Vt = VtB[buf];

#if HAVE_TDM
    if (w == 0) __builtin_amdgcn_s_wait_tensorcnt(0);  // tile j landed in LDS
    __syncthreads();                                   // publish tile j / retire j-1 reads
    if (w == 0 && j + 1 < jend) {                      // DMA tile j+1 under compute(j)
      const int nb = (j + 1) & 1;
      const size_t koff = hb + (size_t)(kb + 32) * HD;
      tdm_load_2d((u32)(uintptr_t)&KhB[nb][0], khg + koff, TDM_FLAGS_K, 64u, (u32)TT, 64u, 32u, 64u);
      tdm_load_2d((u32)(uintptr_t)&KlB[nb][0], klg + koff, TDM_FLAGS_K, 64u, (u32)TT, 64u, 32u, 64u);
      tdm_load_2d((u32)(uintptr_t)&VtB[nb][0], vvg + hb + kb + 32, TDM_FLAGS_V, 32u, (u32)HD, 32u, 64u, (u32)TT);
    }
#else
    {
      const int r = tid >> 2;
      const int c0 = (tid & 3) * 16;
      const size_t base = hb + (size_t)(kb + r) * HD + c0;
#pragma unroll
      for (int i = 0; i < 16; ++i) {
        KhB[buf][r * 80 + c0 + i] = khg[base + i];
        KlB[buf][r * 80 + c0 + i] = klg[base + i];
        VtB[buf][(c0 + i) * 40 + r] = vvg[hb + (size_t)(c0 + i) * TT + kb + r];
      }
    }
    __syncthreads();
#endif

    if (kb <= qmax) {
      v8f s[2] = {};
#pragma unroll
      for (int nt = 0; nt < 2; ++nt) {
#pragma unroll
        for (int ds = 0; ds < 2; ++ds) {
          const bf16_t* pbh = Kh + (nt * 16 + nn) * 80 + ds * 32 + 8 * h2;
          union { v16bf v; uint4 u[2]; } fb;
          fb.u[0] = *(const uint4*)pbh; fb.u[1] = *(const uint4*)(pbh + 16);
          const bf16_t* pbl = Kl + (nt * 16 + nn) * 80 + ds * 32 + 8 * h2;
          union { v16bf v; uint4 u[2]; } fl;
          fl.u[0] = *(const uint4*)pbl; fl.u[1] = *(const uint4*)(pbl + 16);
          s[nt] = wmma_bf16(qfh[ds], fb.v, s[nt]);
          s[nt] = wmma_bf16(qfh[ds], fl.v, s[nt]);
          s[nt] = wmma_bf16(qfl[ds], fb.v, s[nt]);
        }
      }
#pragma unroll
      for (int nt = 0; nt < 2; ++nt) {
#pragma unroll
        for (int r = 0; r < 8; ++r) {
          const int qr = q0 + 16 * w + 8 * h2 + r;
          const int kc = kb + nt * 16 + nn;
          const float sv = s[nt][r] * 0.125f;
          s[nt][r] = (kc <= qr) ? sv : -1e30f;
        }
      }
#pragma unroll
      for (int r = 0; r < 8; ++r) {
        float m = fmaxf(s[0][r], s[1][r]);
#pragma unroll
        for (int off = 1; off < 16; off <<= 1)
          m = fmaxf(m, __shfl_xor(m, off, 32));
        const float nm = fmaxf(runmax[r], m);
        const float alpha = __expf(runmax[r] - nm);
        runmax[r] = nm;
        const float p0 = __expf(s[0][r] - nm);
        const float p1 = __expf(s[1][r] - nm);
        s[0][r] = p0; s[1][r] = p1;
        float rs = p0 + p1;
#pragma unroll
        for (int off = 1; off < 16; off <<= 1)
          rs += __shfl_xor(rs, off, 32);
        lsum[r] = lsum[r] * alpha + rs;
#pragma unroll
        for (int dt = 0; dt < 4; ++dt) O[dt][r] *= alpha;
      }
#pragma unroll
      for (int nt = 0; nt < 2; ++nt)
#pragma unroll
        for (int r = 0; r < 8; ++r)
          Pw[(8 * h2 + r) * 40 + nt * 16 + nn] = (bf16_t)s[nt][r];

      v16bf pf = frag_from_lds_a(Pw, 40);
#pragma unroll
      for (int dt = 0; dt < 4; ++dt) {
        v16bf vf = frag_from_lds_b(Vt + dt * 16 * 40, 40);
        O[dt] = wmma_bf16(pf, vf, O[dt]);
      }
    }
#if !HAVE_TDM
    __syncthreads();
#endif
  }

  float inv[8];
#pragma unroll
  for (int r = 0; r < 8; ++r) inv[r] = 1.0f / lsum[r];
#pragma unroll
  for (int dt = 0; dt < 4; ++dt)
#pragma unroll
    for (int r = 0; r < 8; ++r) {
      const int t = q0 + 16 * w + 8 * h2 + r;
      const size_t row = (size_t)b * TT + t;
      att[row * CE + h * HD + dt * 16 + nn] = O[dt][r] * inv[r];
    }
}

extern "C" void kernel_launch(void* const* d_in, const int* in_sizes, int n_in,
                              void* d_out, int out_size, void* d_ws, size_t ws_size,
                              hipStream_t stream) {
  (void)in_sizes; (void)n_in; (void)out_size; (void)ws_size;
  const float* x    = (const float*)d_in[0];
  const float* Wqkv = (const float*)d_in[1];
  const float* bqkv = (const float*)d_in[2];
  const float* Wout = (const float*)d_in[3];
  const float* bout = (const float*)d_in[4];
  float* out = (float*)d_out;

  const size_t nE = (size_t)BB * NH * TT * HD;
  bf16_t* qh = (bf16_t*)d_ws;
  bf16_t* ql = qh + nE;
  bf16_t* kh = ql + nE;
  bf16_t* kl = kh + nE;
  bf16_t* vv = kl + nE;
  float*  att = (float*)(vv + nE);

  dim3 blk(128);
  gemm_split<0><<<dim3(128, 36), blk, 0, stream>>>(
      x, Wqkv, bqkv, qh, ql, kh, kl, vv, nullptr, CE, 3 * CE);
  attn_flash<<<dim3(TT / 64, BB * NH), blk, 0, stream>>>(qh, ql, kh, kl, vv, att);
  gemm_split<1><<<dim3(128, 12), blk, 0, stream>>>(
      att, Wout, bout, nullptr, nullptr, nullptr, nullptr, nullptr, out, CE, CE);
}
